// Update_19842748907760
// MI455X (gfx1250) — matrix operand order repaired and checked by reading, not compiled, not run on hardware
//
#include <hip/hip_runtime.h>
#include <cstdint>
#include <cstddef>

#define DIM 384
#define NHEAD 8
#define HD 48
#define CORR_DIMC 882
#define NFRAMES 24
#define PPI 96
#define NPATCH (NFRAMES * PPI)      // 2304
#define E_TOT (NPATCH * NFRAMES)    // 55296
#define M_TOK 96
#define NTOK (E_TOT / M_TOK)        // 576

typedef __attribute__((ext_vector_type(16))) __bf16 v16bf;
typedef __attribute__((ext_vector_type(8)))  __bf16 v8bf;
typedef __attribute__((ext_vector_type(4)))  __bf16 v4bf;
typedef __attribute__((ext_vector_type(8)))  float  v8f;

#define BM 128
#define BN 128
#define BK 32
#define LDSK (BK + 8)   // 80B row pitch: keeps 16B alignment for b128 fragment reads

// ---------------------------------------------------------------------------
// Tiled WMMA GEMM:  C[M x N] = act( A[M x K] @ W[K x N] + bias ) (+ Res)
// A/W converted fp32 -> bf16 into LDS; v_wmma_f32_16x16x32_bf16 accumulate.
// PRE:    apply relu to A elements while loading
// ACT:    1 = relu epilogue
// RES:    add residual (read at output location of Res)
// SPLIT:  A is [A | A2] along K (K1 = width of A); K1 multiple of BK required
// FULLK:  K multiple of BK and lda multiple of 4 -> guard-free float4 loads.
//         FULLK==0 assumes lda even (float2 fast path); the single K-remainder
//         tile uses branch-free clamped loads + select-zero.
// M (grid.y*BM) and N (grid.x*BN) must be multiples of the tile (true here).
// ---------------------------------------------------------------------------
template<int PRE, int ACT, int RES, int SPLIT, int FULLK>
__global__ __launch_bounds__(256) void gemm_wmma(
    const float* __restrict__ A, int lda,
    const float* __restrict__ A2, int lda2, int K1,
    const float* __restrict__ W, const float* __restrict__ bias,
    const float* __restrict__ Res, float* __restrict__ C,
    int N, int K)
{
  __shared__ __align__(16) __bf16 As[BM][LDSK];
  __shared__ __align__(16) __bf16 Bs[BN][LDSK];

  const int tid  = threadIdx.x;
  const int lane = tid & 31;
  const int wave = tid >> 5;
  const int wm   = wave & 3;     // 4 wave-rows of 32
  const int wn   = wave >> 2;    // 2 wave-cols of 64
  const int brow = blockIdx.y * BM;
  const int bcol = blockIdx.x * BN;

  v8f acc[2][4];
  #pragma unroll
  for (int i = 0; i < 2; ++i)
    #pragma unroll
    for (int j = 0; j < 4; ++j)
      acc[i][j] = (v8f){0.f,0.f,0.f,0.f,0.f,0.f,0.f,0.f};

  // A loader mapping: 2 threads per row, 16 contiguous K each
  const int arow = tid >> 1;
  const int ac0  = (tid & 1) * 16;
  // B loader mapping: each thread owns a 4k x 4n block of W
  const int bkq  = (tid >> 5) * 4;     // 0..28
  const int bn0  = (tid & 31) * 4;     // 0..124

  for (int kb = 0; kb < K; kb += BK) {
    const float* Asrc = A; int asl = lda; int kloc = kb; int klim = SPLIT ? K1 : K;
    if (SPLIT && kb >= K1) { Asrc = A2; asl = lda2; kloc = kb - K1; klim = K - K1; }
    const int kmax = klim - kloc;     // valid K columns in this tile source

    { // ---- A tile -> LDS (bf16), vectorized ----
      const float* ap = Asrc + (size_t)(brow + arow) * (size_t)asl + kloc + ac0;
      float va[16];
      if (FULLK) {
        float4 v0 = *(const float4*)(ap + 0);
        float4 v1 = *(const float4*)(ap + 4);
        float4 v2 = *(const float4*)(ap + 8);
        float4 v3 = *(const float4*)(ap + 12);
        va[0]=v0.x; va[1]=v0.y; va[2]=v0.z; va[3]=v0.w;
        va[4]=v1.x; va[5]=v1.y; va[6]=v1.z; va[7]=v1.w;
        va[8]=v2.x; va[9]=v2.y; va[10]=v2.z; va[11]=v2.w;
        va[12]=v3.x; va[13]=v3.y; va[14]=v3.z; va[15]=v3.w;
      } else if (kmax >= BK) {        // uniform fast path (lda even -> float2 ok)
        #pragma unroll
        for (int i = 0; i < 8; ++i) {
          float2 u = *(const float2*)(ap + 2 * i);
          va[2*i] = u.x; va[2*i + 1] = u.y;
        }
      } else {
        // Remainder tile (corr GEMM only): branch-free clamped load + select.
        // ap[0] is always inside the current row (kloc+ac0 <= K-2), so the
        // clamped address is legal; invalid lanes are zeroed with cndmask.
        #pragma unroll
        for (int c = 0; c < 16; ++c) {
          bool ok = (ac0 + c) < kmax;
          int  cc = ok ? c : 0;
          float v = ap[cc];
          va[c] = ok ? v : 0.f;
        }
      }
      v8bf p0, p1;
      #pragma unroll
      for (int c = 0; c < 8; ++c) {
        float lo = va[c], hi = va[8 + c];
        if (PRE) { lo = fmaxf(lo, 0.f); hi = fmaxf(hi, 0.f); }
        p0[c] = (__bf16)lo;
        p1[c] = (__bf16)hi;
      }
      *(v8bf*)&As[arow][ac0]     = p0;
      *(v8bf*)&As[arow][ac0 + 8] = p1;
    }

    { // ---- W tile -> LDS transposed [n][k], vectorized ----
      float rr[4][4];
      if (FULLK) {
        #pragma unroll
        for (int j = 0; j < 4; ++j) {
          float4 v = *(const float4*)(W + (size_t)(kb + bkq + j) * (size_t)N + bcol + bn0);
          rr[j][0] = v.x; rr[j][1] = v.y; rr[j][2] = v.z; rr[j][3] = v.w;
        }
      } else {
        // Branch-free: clamp row index to K-1 (always a valid W row), then
        // zero the lanes whose true row is out of range.
        #pragma unroll
        for (int j = 0; j < 4; ++j) {
          int kg = kb + bkq + j;
          bool ok = kg < K;
          int  kc = ok ? kg : (K - 1);
          float4 v = *(const float4*)(W + (size_t)kc * (size_t)N + bcol + bn0);
          rr[j][0] = ok ? v.x : 0.f;
          rr[j][1] = ok ? v.y : 0.f;
          rr[j][2] = ok ? v.z : 0.f;
          rr[j][3] = ok ? v.w : 0.f;
        }
      }
      #pragma unroll
      for (int n = 0; n < 4; ++n) {
        v4bf p;
        p[0] = (__bf16)rr[0][n];
        p[1] = (__bf16)rr[1][n];
        p[2] = (__bf16)rr[2][n];
        p[3] = (__bf16)rr[3][n];
        *(v4bf*)&Bs[bn0 + n][bkq] = p;   // 4 contiguous K halves -> ds_store_b64
      }
    }
    __syncthreads();

    {
      const int l15   = lane & 15;
      const int ahalf = (lane >> 4) * 8;   // A: lanes 16-31 own K+8 / K+24 halves
      const int bhalf = (lane >> 4) * 8;   // B: lanes 16-31 own K+8 / K+24 halves
      v16bf afr[2]; v16bf bfr[4];
      #pragma unroll
      for (int tm = 0; tm < 2; ++tm) {
        int r = wm * 32 + tm * 16 + l15;
        v8bf lo = *(const v8bf*)&As[r][ahalf];
        v8bf hi = *(const v8bf*)&As[r][ahalf + 16];
        afr[tm] = __builtin_shufflevector(lo, hi, 0,1,2,3,4,5,6,7,8,9,10,11,12,13,14,15);
      }
      #pragma unroll
      for (int tn = 0; tn < 4; ++tn) {
        int n = wn * 64 + tn * 16 + l15;
        v8bf lo = *(const v8bf*)&Bs[n][bhalf];
        v8bf hi = *(const v8bf*)&Bs[n][bhalf + 16];
        bfr[tn] = __builtin_shufflevector(lo, hi, 0,1,2,3,4,5,6,7,8,9,10,11,12,13,14,15);
      }
      #pragma unroll
      for (int tm = 0; tm < 2; ++tm)
        #pragma unroll
        for (int tn = 0; tn < 4; ++tn)
          acc[tm][tn] = __builtin_amdgcn_wmma_f32_16x16x32_bf16(
              false, afr[tm], false, bfr[tn], (short)0, acc[tm][tn], false, false);
    }
    __syncthreads();
  }

  // Epilogue.  C/D layout: lane l -> N = l&15; VGPR r -> M = r + 8*(l>>4).
  const int l15  = lane & 15;
  const int rofs = (lane >> 4) << 3;
  #pragma unroll
  for (int tn = 0; tn < 4; ++tn) {
    int col = bcol + wn * 64 + tn * 16 + l15;
    float bv = bias[col];
    #pragma unroll
    for (int tm = 0; tm < 2; ++tm) {
      int rbase = brow + wm * 32 + tm * 16 + rofs;
      #pragma unroll
      for (int r = 0; r < 8; ++r) {
        float v = acc[tm][tn][r] + bv;
        if (ACT == 1) v = fmaxf(v, 0.f);
        size_t off = (size_t)(rbase + r) * (size_t)N + col;
        if (RES) v += Res[off];
        C[off] = v;
      }
    }
  }
}

// ---------------------------------------------------------------------------
// LayerNorm over last dim (384). Optional 3-way fused add and gather index.
// out[row] = LN(x0[idx?idx[row]:row] (+ x1[row]) (+ x2[row])) * g + b
// ---------------------------------------------------------------------------
__global__ __launch_bounds__(DIM) void ln_kernel(
    const float* __restrict__ x0, const float* __restrict__ x1,
    const float* __restrict__ x2, const int* __restrict__ idx,
    const float* __restrict__ g, const float* __restrict__ b,
    float* __restrict__ out)
{
  __shared__ float red[DIM];
  const int row = blockIdx.x;
  const int d = threadIdx.x;
  const int sr = idx ? idx[row] : row;
  float v = x0[(size_t)sr * DIM + d];
  if (x1) v += x1[(size_t)row * DIM + d];
  if (x2) v += x2[(size_t)row * DIM + d];
  red[d] = v;
  __syncthreads();
  for (int s = 192; s >= 3; s >>= 1) { if (d < s) red[d] += red[d + s]; __syncthreads(); }
  float mu = (red[0] + red[1] + red[2]) * (1.0f / DIM);
  __syncthreads();
  float dv = v - mu;
  red[d] = dv * dv;
  __syncthreads();
  for (int s = 192; s >= 3; s >>= 1) { if (d < s) red[d] += red[d + s]; __syncthreads(); }
  float var = (red[0] + red[1] + red[2]) * (1.0f / DIM);
  float rstd = rsqrtf(var + 1e-3f);
  out[(size_t)row * DIM + d] = dv * rstd * g[d] + b[d];
}

// dst[row] = (gi[row] >= 0) ? src[gi[row]] : 0
__global__ __launch_bounds__(DIM) void gather_mask_kernel(
    const float* __restrict__ src, const int* __restrict__ gi,
    float* __restrict__ dst)
{
  const int row = blockIdx.x, d = threadIdx.x;
  const int s = gi[row];
  dst[(size_t)row * DIM + d] = (s >= 0) ? src[(size_t)s * DIM + d] : 0.f;
}

// out[row] = a[row] + h[kk[row]]
__global__ __launch_bounds__(DIM) void add_gather_kernel(
    const float* __restrict__ a, const float* __restrict__ h,
    const int* __restrict__ kk, float* __restrict__ out)
{
  const int row = blockIdx.x, d = threadIdx.x;
  out[(size_t)row * DIM + d] =
      a[(size_t)row * DIM + d] + h[(size_t)kk[row] * DIM + d];
}

// Segment softmax aggregation; segments are contiguous runs of NFRAMES rows.
__global__ __launch_bounds__(256) void softagg_kernel(
    const float* __restrict__ g, const float* __restrict__ f,
    float* __restrict__ y)
{
  const int idx = blockIdx.x * 256 + threadIdx.x;
  if (idx >= NPATCH * DIM) return;
  const int p = idx / DIM, d = idx - p * DIM;
  const size_t base = (size_t)p * NFRAMES * DIM + d;
  float m = -3.4e38f;
  for (int t = 0; t < NFRAMES; ++t) m = fmaxf(m, g[base + (size_t)t * DIM]);
  float den = 0.f, num = 0.f;
  for (int t = 0; t < NFRAMES; ++t) {
    float e = __expf(g[base + (size_t)t * DIM] - m);
    den += e;
    num += e * f[base + (size_t)t * DIM];
  }
  y[(size_t)p * DIM + d] = num / den;
}

// ---------------------------------------------------------------------------
// Gated linear attention core for one (token-group n, head h).
// RoPE + (elu+1) feature map, KV = K^T V (48x48), out = (Q KV) / (Q.Ksum).
// Out may alias Qb: all global reads complete before the first barrier,
// writes happen after; each block owns disjoint (rows x head-cols).
// ---------------------------------------------------------------------------
__global__ __launch_bounds__(96) void attn_kernel(
    const float* __restrict__ Qb, const float* __restrict__ Kb,
    const float* __restrict__ Vb, const int* __restrict__ flat,
    const float* __restrict__ posenc, float* __restrict__ Out)
{
  __shared__ float Ks[M_TOK][HD];
  __shared__ float Vs[M_TOK][HD];
  __shared__ float KVs[HD][HD + 1];
  __shared__ float Ksum[HD];

  const int n = blockIdx.x;
  const int h = blockIdx.y;
  const int l = threadIdx.x;
  const int rowe = n * M_TOK + l;
  const size_t base = (size_t)rowe * DIM + h * HD;
  const int pi = flat[rowe];
  const float* cosb = posenc + (size_t)pi * HD;
  const float* sinb = posenc + (size_t)E_TOT * HD + (size_t)pi * HD;

  float Qr[HD];
  #pragma unroll
  for (int d = 0; d < HD; ++d) {
    float cv = cosb[d], sv = sinb[d];
    float qv = Qb[base + d];
    float kv = Kb[base + d];
    float qr = (d < HD / 2) ? -Qb[base + d + HD / 2] : Qb[base + d - HD / 2];
    float kr = (d < HD / 2) ? -Kb[base + d + HD / 2] : Kb[base + d - HD / 2];
    qv = qv * cv + qr * sv;
    kv = kv * cv + kr * sv;
    Qr[d]    = (qv > 0.f) ? qv + 1.f : __expf(qv);   // elu(x)+1
    Ks[l][d] = (kv > 0.f) ? kv + 1.f : __expf(kv);
    Vs[l][d] = Vb[base + d];
  }
  __syncthreads();

  #pragma unroll 1
  for (int m = 0; m < 24; ++m) {              // 48*48 / 96 threads = 24 each
    int idx = l * 24 + m;
    int dd = idx / HD, vv = idx % HD;
    float s = 0.f;
    for (int t = 0; t < M_TOK; ++t) s += Ks[t][dd] * Vs[t][vv];
    KVs[dd][vv] = s;
  }
  if (l < HD) {
    float s = 0.f;
    for (int t = 0; t < M_TOK; ++t) s += Ks[t][l];
    Ksum[l] = s;
  }
  __syncthreads();

  float z = 0.f;
  #pragma unroll
  for (int d = 0; d < HD; ++d) z += Qr[d] * Ksum[d];
  z = 1.f / (z + 1e-6f);
  #pragma unroll 1
  for (int vv = 0; vv < HD; ++vv) {
    float s = 0.f;
    #pragma unroll
    for (int d = 0; d < HD; ++d) s += Qr[d] * KVs[d][vv];
    Out[base + vv] = s * z;
  }
}

// outnet[flat[i]] = X[i] + sigmoid(gl[i]) * msg[i]
__global__ __launch_bounds__(DIM) void combine_scatter_kernel(
    const float* __restrict__ X, const float* __restrict__ msg,
    const float* __restrict__ gl, const int* __restrict__ flat,
    float* __restrict__ outnet)
{
  const int i = blockIdx.x, d = threadIdx.x;
  const size_t off = (size_t)i * DIM + d;
  const float gv = 1.f / (1.f + __expf(-gl[off]));
  outnet[(size_t)flat[i] * DIM + d] = X[off] + gv * msg[off];
}

// d = relu(net) @ Wd + bd ; w = sigmoid(relu(net) @ Ww + bw)   (N=2 heads)
__global__ __launch_bounds__(256) void heads_kernel(
    const float* __restrict__ net, const float* __restrict__ Wd,
    const float* __restrict__ bd, const float* __restrict__ Ww,
    const float* __restrict__ bw, float* __restrict__ dout,
    float* __restrict__ wout)
{
  const int e = blockIdx.x * 256 + threadIdx.x;
  if (e >= E_TOT) return;
  const float* nr = net + (size_t)e * DIM;
  float a0 = 0.f, a1 = 0.f, c0 = 0.f, c1 = 0.f;
  #pragma unroll 4
  for (int j = 0; j < DIM; ++j) {
    float v = fmaxf(nr[j], 0.f);
    a0 += v * Wd[j * 2 + 0]; a1 += v * Wd[j * 2 + 1];
    c0 += v * Ww[j * 2 + 0]; c1 += v * Ww[j * 2 + 1];
  }
  dout[e * 2 + 0] = a0 + bd[0];
  dout[e * 2 + 1] = a1 + bd[1];
  wout[e * 2 + 0] = 1.f / (1.f + __expf(-(c0 + bw[0])));
  wout[e * 2 + 1] = 1.f / (1.f + __expf(-(c1 + bw[1])));
}

// ---------------------------------------------------------------------------
extern "C" void kernel_launch(void* const* d_in, const int* in_sizes, int n_in,
                              void* d_out, int out_size, void* d_ws, size_t ws_size,
                              hipStream_t stream)
{
  (void)in_sizes; (void)n_in; (void)out_size; (void)ws_size;

  const float* net    = (const float*)d_in[0];
  const float* inp    = (const float*)d_in[1];
  const float* corr   = (const float*)d_in[2];
  const float* posenc = (const float*)d_in[4];
  const int*   kk     = (const int*)d_in[7];
  const int*   ix     = (const int*)d_in[8];
  const int*   jx     = (const int*)d_in[9];
  const int*   flat   = (const int*)d_in[10];   // ij_ind flattened (perm of E)
  const float* Wc1 = (const float*)d_in[11]; const float* bc1 = (const float*)d_in[12];
  const float* Wc2 = (const float*)d_in[13]; const float* bc2 = (const float*)d_in[14];
  const float* g_lnc = (const float*)d_in[15]; const float* b_lnc = (const float*)d_in[16];
  const float* Wc3 = (const float*)d_in[17]; const float* bc3 = (const float*)d_in[18];
  const float* g_norm = (const float*)d_in[19]; const float* b_norm = (const float*)d_in[20];
  const float* W1a = (const float*)d_in[21]; const float* b1a = (const float*)d_in[22];
  const float* W1b = (const float*)d_in[23]; const float* b1b = (const float*)d_in[24];
  const float* W2a = (const float*)d_in[25]; const float* b2a = (const float*)d_in[26];
  const float* W2b = (const float*)d_in[27]; const float* b2b = (const float*)d_in[28];
  const float* Wf  = (const float*)d_in[29]; const float* bf  = (const float*)d_in[30];
  const float* Wg  = (const float*)d_in[31]; const float* bg  = (const float*)d_in[32];
  const float* Wh  = (const float*)d_in[33]; const float* bh  = (const float*)d_in[34];
  const float* g_norm2 = (const float*)d_in[35]; const float* b_norm2 = (const float*)d_in[36];
  const float* Wq  = (const float*)d_in[37]; const float* bq  = (const float*)d_in[38];
  const float* Wk  = (const float*)d_in[39]; const float* bk  = (const float*)d_in[40];
  const float* Wv  = (const float*)d_in[41]; const float* bv  = (const float*)d_in[42];
  const float* Wm  = (const float*)d_in[43]; const float* bm  = (const float*)d_in[44];
  const float* Wgt = (const float*)d_in[45]; const float* bgt = (const float*)d_in[46];
  const float* Wd  = (const float*)d_in[47]; const float* bd  = (const float*)d_in[48];
  const float* Ww  = (const float*)d_in[49]; const float* bw  = (const float*)d_in[50];

  // Scratch layout: 4 big E x DIM slabs + 2 NPATCH x DIM slabs (~347 MB).
  const size_t BIG = (size_t)E_TOT * DIM;
  float* B0 = (float*)d_ws;
  float* B1 = B0 + BIG;
  float* B2 = B1 + BIG;
  float* B3 = B2 + BIG;
  float* S0 = B3 + BIG;
  float* S1 = S0 + (size_t)NPATCH * DIM;

  float* out_net = (float*)d_out;
  float* out_d   = out_net + BIG;
  float* out_w   = out_d + (size_t)E_TOT * 2;

  const dim3 blk(256);
  const dim3 gE(DIM / BN, E_TOT / BM);    // (3, 432)
  const dim3 gP(DIM / BN, NPATCH / BM);   // (3, 18)

  // ---- correlation encoder ----
  gemm_wmma<0,1,0,0,0><<<gE, blk, 0, stream>>>(corr, CORR_DIMC, nullptr, 0, 0,
      Wc1, bc1, nullptr, B0, DIM, CORR_DIMC);                       // c1 = relu(corr@Wc1+b)
  gemm_wmma<0,0,0,0,1><<<gE, blk, 0, stream>>>(B0, DIM, nullptr, 0, 0,
      Wc2, bc2, nullptr, B1, DIM, DIM);                             // t2 = c1@Wc2+b
  ln_kernel<<<E_TOT, DIM, 0, stream>>>(B1, nullptr, nullptr, nullptr,
      g_lnc, b_lnc, B0);                                            // c2 = LN(t2)
  gemm_wmma<1,0,0,0,1><<<gE, blk, 0, stream>>>(B0, DIM, nullptr, 0, 0,
      Wc3, bc3, nullptr, B1, DIM, DIM);                             // c3 = relu(c2)@Wc3+b
  ln_kernel<<<E_TOT, DIM, 0, stream>>>(net, inp, B1, nullptr,
      g_norm, b_norm, B0);                                          // net0 = LN(net+inp+c3)

  // ---- temporal MLPs (ix then jx) ----
  gather_mask_kernel<<<E_TOT, DIM, 0, stream>>>(B0, ix, B1);
  gemm_wmma<0,1,0,0,1><<<gE, blk, 0, stream>>>(B1, DIM, nullptr, 0, 0,
      W1a, b1a, nullptr, B2, DIM, DIM);
  gemm_wmma<0,0,1,0,1><<<gE, blk, 0, stream>>>(B2, DIM, nullptr, 0, 0,
      W1b, b1b, B0, B1, DIM, DIM);                                  // net1 = net0 + mlp
  gather_mask_kernel<<<E_TOT, DIM, 0, stream>>>(B1, jx, B0);
  gemm_wmma<0,1,0,0,1><<<gE, blk, 0, stream>>>(B0, DIM, nullptr, 0, 0,
      W2a, b2a, nullptr, B2, DIM, DIM);
  gemm_wmma<0,0,1,0,1><<<gE, blk, 0, stream>>>(B2, DIM, nullptr, 0, 0,
      W2b, b2b, B1, B0, DIM, DIM);                                  // net2 = net1 + mlp

  // ---- soft aggregation over patches ----
  gemm_wmma<0,0,0,0,1><<<gE, blk, 0, stream>>>(B0, DIM, nullptr, 0, 0,
      Wg, bg, nullptr, B1, DIM, DIM);                               // g
  gemm_wmma<0,0,0,0,1><<<gE, blk, 0, stream>>>(B0, DIM, nullptr, 0, 0,
      Wf, bf, nullptr, B2, DIM, DIM);                               // f
  softagg_kernel<<<(NPATCH * DIM + 255) / 256, 256, 0, stream>>>(B1, B2, S0);
  gemm_wmma<0,0,0,0,1><<<gP, blk, 0, stream>>>(S0, DIM, nullptr, 0, 0,
      Wh, bh, nullptr, S1, DIM, DIM);                               // h = y@Wh+b
  add_gather_kernel<<<E_TOT, DIM, 0, stream>>>(B0, S1, kk, B1);     // net3 = net2 + h[kk]

  // ---- token attention block ----
  ln_kernel<<<E_TOT, DIM, 0, stream>>>(B1, nullptr, nullptr, flat,
      g_norm2, b_norm2, B0);                                        // X = LN(net3[flat])
  gemm_wmma<0,0,0,0,1><<<gE, blk, 0, stream>>>(B0, DIM, nullptr, 0, 0,
      Wq, bq, nullptr, B2, DIM, DIM);                               // Q
  gemm_wmma<0,0,0,0,1><<<gE, blk, 0, stream>>>(B0, DIM, nullptr, 0, 0,
      Wk, bk, nullptr, B3, DIM, DIM);                               // K
  gemm_wmma<0,0,0,0,1><<<gE, blk, 0, stream>>>(B0, DIM, nullptr, 0, 0,
      Wv, bv, nullptr, B1, DIM, DIM);                               // V
  attn_kernel<<<dim3(NTOK, NHEAD), 96, 0, stream>>>(B2, B3, B1, flat,
      posenc, B2);                                                  // attn out (in-place)
  gemm_wmma<0,0,0,0,1><<<gE, blk, 0, stream>>>(B2, DIM, nullptr, 0, 0,
      Wm, bm, nullptr, B1, DIM, DIM);                               // msg
  gemm_wmma<0,0,0,1,1><<<gE, blk, 0, stream>>>(B0, DIM, B1, DIM, DIM,
      Wgt, bgt, nullptr, B3, DIM, 2 * DIM);                         // gate pre-sigmoid
  combine_scatter_kernel<<<E_TOT, DIM, 0, stream>>>(B0, B1, B3, flat,
      out_net);                                                     // net4 scattered

  // ---- output heads ----
  heads_kernel<<<(E_TOT + 255) / 256, 256, 0, stream>>>(out_net,
      Wd, bd, Ww, bw, out_d, out_w);
}
